// LT_FN_1640677507425
// MI455X (gfx1250) — matrix operand order repaired and checked by reading, not compiled, not run on hardware
//
#include <hip/hip_runtime.h>
#include <hip/hip_bf16.h>
#include <stdint.h>
#include <stddef.h>

typedef _Float16 f16;
typedef __attribute__((ext_vector_type(16))) _Float16 v16h;
typedef __attribute__((ext_vector_type(8)))  _Float16 v8h;
typedef __attribute__((ext_vector_type(8)))  float    v8f;

#define EP_RELU   1
#define EP_BN     2
#define EP_RESID  4
#define EP_CSCALE 8

static inline int cdiv(long long a, long long b){ return (int)((a + b - 1) / b); }

// ---------------------------------------------------------------------------
// Elementwise / data-movement kernels
// ---------------------------------------------------------------------------

// x [B,C,H,W] f32 -> feature matrix [B*H*W, C] f16
__global__ void k_nchw_to_feat(const float* __restrict__ x, f16* __restrict__ f,
                               int B, int C, int HW){
  int i = blockIdx.x*blockDim.x + threadIdx.x;
  int tot = B*C*HW;
  if (i >= tot) return;
  int p = i % HW; int c = (i/HW) % C; int b = i/(HW*C);
  f[(b*HW + p)*C + c] = (f16)x[i];
}

// Fold Linear bias + BatchNorm(eval): y = acc*scale + shift ; fill = relu(bn shift only)
__global__ void k_bnprep(const float* __restrict__ b, const float* __restrict__ gamma,
                         const float* __restrict__ beta, const float* __restrict__ mean,
                         const float* __restrict__ var,
                         float* __restrict__ scale, float* __restrict__ shift,
                         float* __restrict__ fill, int C){
  int c = blockIdx.x*blockDim.x + threadIdx.x;
  if (c >= C) return;
  float s = gamma[c] * rsqrtf(var[c] + 1e-5f);
  float bn_shift = beta[c] - mean[c]*s;   // zero-padded slots get only BN affine
  scale[c] = s;
  shift[c] = b[c]*s + bn_shift;           // computed slots get bias too
  fill[c]  = fmaxf(bn_shift, 0.f);
}

// write relu(bn shift) into the never-visited spatial slots (stride>1 units)
__global__ void k_fill(f16* __restrict__ feat, const float* __restrict__ fill,
                       int B, int C, int Pfull, int Pfilled){
  int per_b = (Pfull - Pfilled)*C;
  int i = blockIdx.x*blockDim.x + threadIdx.x;
  if (i >= B*per_b) return;
  int b = i/per_b; int r = i%per_b;
  int p = Pfilled + r/C; int c = r%C;
  feat[((long long)b*Pfull + p)*C + c] = (f16)fill[c];
}

// im2col (reference ordering: row = b*P1*P1 + i*P1+j, col = ci*k*k + dy*k+dx),
// zero-padded out to Kpad (multiple of 32) so the GEMM needs no K guards.
__global__ void k_im2col(const f16* __restrict__ feat, f16* __restrict__ A,
                         int B, int Cin, int H, int W, int k, int s, int pad,
                         int P1, int Kreal, int Kpad){
  long long i = (long long)blockIdx.x*blockDim.x + threadIdx.x;
  long long tot = (long long)B*P1*P1*Kpad;
  if (i >= tot) return;
  int col = (int)(i % Kpad); long long row = i / Kpad;
  f16 v = (f16)0.f;
  if (col < Kreal) {
    int ci = col/(k*k); int kk = col%(k*k); int dy = kk/k; int dx = kk%k;
    int p = (int)(row % (P1*P1)); int b = (int)(row/(P1*P1));
    int pi = p/P1, pj = p%P1;
    int y = pi*s - pad + dy;
    int x = pj*s - pad + dx;
    if (y >= 0 && y < H && x >= 0 && x < W)
      v = feat[((long long)b*H*W + y*W + x)*Cin + ci];
  }
  A[row*Kpad + col] = v;
}

__global__ void k_addrelu(const f16* __restrict__ a, const f16* __restrict__ b,
                          f16* __restrict__ o, int n){
  int i = blockIdx.x*blockDim.x + threadIdx.x;
  if (i < n) o[i] = (f16)fmaxf((float)a[i] + (float)b[i], 0.f);
}

// ---------------------------------------------------------------------------
// B-matrix packing, pair-interleaved WMMA operand layout, contiguous k per half-wave:
//   Bp[ (((z*(N/32) + tjj)*K32 + kc)*1024) + half*512 + lane*16 + e ]
//   n = (tjj*2 + half)*16 + (lane&15),  kk = kc*32 + (lane>>4)*16 + e
// One wave consumes both tiles of a pair each k-step as ONE contiguous 2KB run
// (four b128 loads from a single base at offsets 0/16/1024/1040 bytes).
// ---------------------------------------------------------------------------

__global__ void k_pack_dense(const float* __restrict__ src, f16* __restrict__ dst,
                             int K, int N, int K32){
  long long i = (long long)blockIdx.x*blockDim.x + threadIdx.x;
  long long tot = (long long)(N >> 4)*K32*512;
  if (i >= tot) return;
  int e = (int)(i & 15); int lane = (int)((i >> 4) & 31);
  int half = (int)((i >> 9) & 1);
  long long r = i >> 10;
  int kc = (int)(r % K32); int tjj = (int)(r / K32);
  int n  = (tjj*2 + half)*16 + (lane & 15);
  int kk = kc*32 + (lane >> 4)*16 + e;
  dst[i] = (kk < K) ? (f16)src[(long long)kk*N + n] : (f16)0.f;
}

// conv weights W[Cin][Cout][KK] -> packed, logical B row index kk = ci*KK + k9
__global__ void k_pack_conv(const float* __restrict__ W, f16* __restrict__ dst,
                            int Cin, int Cout, int KK, int K32){
  long long i = (long long)blockIdx.x*blockDim.x + threadIdx.x;
  long long tot = (long long)(Cout >> 4)*K32*512;
  if (i >= tot) return;
  int e = (int)(i & 15); int lane = (int)((i >> 4) & 31);
  int half = (int)((i >> 9) & 1);
  long long r = i >> 10;
  int kc = (int)(r % K32); int tjj = (int)(r / K32);
  int n  = (tjj*2 + half)*16 + (lane & 15);
  int kk = kc*32 + (lane >> 4)*16 + e;
  f16 v = (f16)0.f;
  if (kk < Cin*KK) {
    int ci = kk/KK, k9 = kk%KK;
    v = (f16)W[((long long)ci*Cout + n)*KK + k9];
  }
  dst[i] = v;
}

// pack an f16 activation sub-matrix (batched, optional transpose) -> packed B
// trans==0: B[kk][n] = src[off + kk*lds + n]   (e.g. V blocks)
// trans==1: B[kk][n] = src[off + n*lds + kk]   (e.g. K^T blocks)
__global__ void k_pack_f16(const f16* __restrict__ src, f16* __restrict__ dst,
                           int K, int N, int K32, int lds, int trans,
                           int Z, int zdiv, long long so, long long si){
  long long i = (long long)blockIdx.x*blockDim.x + threadIdx.x;
  long long per_z = (long long)(N >> 4)*K32*512;
  if (i >= per_z*Z) return;
  int z = (int)(i / per_z); long long r0 = i % per_z;
  long long off = (long long)(z/zdiv)*so + (long long)(z%zdiv)*si;
  int e = (int)(r0 & 15); int lane = (int)((r0 >> 4) & 31);
  int half = (int)((r0 >> 9) & 1);
  long long r = r0 >> 10;
  int kc = (int)(r % K32); int tjj = (int)(r / K32);
  int n  = (tjj*2 + half)*16 + (lane & 15);
  int kk = kc*32 + (lane >> 4)*16 + e;
  f16 v = (f16)0.f;
  if (kk < K)
    v = trans ? src[off + (long long)n*lds + kk]
              : src[off + (long long)kk*lds + n];
  dst[i] = v;
}

// ---------------------------------------------------------------------------
// Batched WMMA GEMM, pair-packed B, 16x32 tile per wave, software-pipelined:
// chunk k+1's 2 A-b128s + 4 B-b128s (one contiguous clause) are issued before
// the WMMAs of chunk k, so loads stay in flight behind the matrix ops.
// ---------------------------------------------------------------------------
__global__ __launch_bounds__(256)
void k_gemm(const f16* __restrict__ A, const f16* __restrict__ Bp,
            float* __restrict__ Cf, f16* __restrict__ Ch,
            const f16* __restrict__ resid,
            const float* __restrict__ scale, const float* __restrict__ shift,
            int M, int N, int K32, int lda, int ldc,
            int flags, float cscale, int rowsPerB, int zdiv,
            long long sAo, long long sAi, long long sCo, long long sCi)
{
  int z = blockIdx.z;
  A += (long long)(z/zdiv)*sAo + (long long)(z%zdiv)*sAi;
  long long offC = (long long)(z/zdiv)*sCo + (long long)(z%zdiv)*sCi;

  int lane = threadIdx.x & 31, wid = threadIdx.x >> 5;
  int tm = M >> 4, tn = N >> 4, tnn = tn >> 1;
  int tile = blockIdx.x*8 + wid;
  if (tile >= tm*tnn) return;
  int ti = tile % tm, tjj = tile / tm, tj = tjj*2;
  int mrow = ti*16 + (lane & 15);
  int aKb  = (lane >> 4)*16;         // A: hi half-wave holds K+16..31

  const v8h* ap = (const v8h*)(A + (long long)mrow*lda + aKb);
  const f16* bbase = Bp + ((long long)z*tnn + tjj)*(long long)K32*1024 + lane*16;
  const v8h* bp = (const v8h*)bbase;   // tile0 at +0, tile1 at +512 halves (= +64 v8h)

  v8f acc0 = {}, acc1 = {};

  // prologue: load chunk 0
  v8h ca0 = ap[0], ca1 = ap[1];
  v8h cb00 = bp[0], cb01 = bp[1];
  v8h cb10 = bp[64], cb11 = bp[65];

  for (int kc = 0; kc < K32 - 1; ++kc) {
    ap += 4; bp += 128;   // A: +32 halves ; B: +1024 halves (one pair-chunk)
    // prefetch chunk kc+1 (issued before the WMMAs that consume chunk kc)
    v8h na0 = ap[0], na1 = ap[1];
    v8h nb00 = bp[0], nb01 = bp[1];
    v8h nb10 = bp[64], nb11 = bp[65];

    v16h af  = __builtin_shufflevector(ca0,  ca1,  0,1,2,3,4,5,6,7,8,9,10,11,12,13,14,15);
    v16h bf0 = __builtin_shufflevector(cb00, cb01, 0,1,2,3,4,5,6,7,8,9,10,11,12,13,14,15);
    v16h bf1 = __builtin_shufflevector(cb10, cb11, 0,1,2,3,4,5,6,7,8,9,10,11,12,13,14,15);
    acc0 = __builtin_amdgcn_wmma_f32_16x16x32_f16(false, af, false, bf0,
                                                  (short)0, acc0, false, false);
    acc1 = __builtin_amdgcn_wmma_f32_16x16x32_f16(false, af, false, bf1,
                                                  (short)0, acc1, false, false);
    ca0 = na0; ca1 = na1;
    cb00 = nb00; cb01 = nb01; cb10 = nb10; cb11 = nb11;
  }
  // pipeline epilogue: compute the final chunk
  {
    v16h af  = __builtin_shufflevector(ca0,  ca1,  0,1,2,3,4,5,6,7,8,9,10,11,12,13,14,15);
    v16h bf0 = __builtin_shufflevector(cb00, cb01, 0,1,2,3,4,5,6,7,8,9,10,11,12,13,14,15);
    v16h bf1 = __builtin_shufflevector(cb10, cb11, 0,1,2,3,4,5,6,7,8,9,10,11,12,13,14,15);
    acc0 = __builtin_amdgcn_wmma_f32_16x16x32_f16(false, af, false, bf0,
                                                  (short)0, acc0, false, false);
    acc1 = __builtin_amdgcn_wmma_f32_16x16x32_f16(false, af, false, bf1,
                                                  (short)0, acc1, false, false);
  }

#pragma unroll
  for (int t = 0; t < 2; ++t) {
    int ncol = (tj + t)*16 + (lane & 15);
    float sc = 1.f, sh = 0.f;
    if (flags & EP_BN) { sc = scale[ncol]; sh = shift[ncol]; }
#pragma unroll
    for (int v = 0; v < 8; ++v) {
      int m = ti*16 + (lane >> 4)*8 + v;  // C layout: VGPR v -> row v (+8 hi lanes)
      float y = t ? acc1[v] : acc0[v];
      if (flags & EP_CSCALE) y *= cscale;
      y = y*sc + sh;
      long long mo = m;
      if (rowsPerB > 0) mo = (long long)(m / rowsPerB)*1024 + (m % rowsPerB);
      long long cix = offC + mo*(long long)ldc + ncol;
      if (flags & EP_RESID) y += (float)resid[cix];
      if (flags & EP_RELU)  y = fmaxf(y, 0.f);
      if (Cf) Cf[cix] = y;
      if (Ch) Ch[cix] = (f16)y;
    }
  }
}

// ---------------------------------------------------------------------------
// Softmax (in-place on f16 rows), LayerNorm, mean-pool, head
// ---------------------------------------------------------------------------

__global__ __launch_bounds__(256)
void k_softmax(f16* __restrict__ S, int n){
  long long row = blockIdx.x;
  f16* r = S + row*(long long)n;
  __shared__ float red[256];
  int t = threadIdx.x;
  float mx = -1e30f;
  for (int j = t; j < n; j += 256) mx = fmaxf(mx, (float)r[j]);
  red[t] = mx; __syncthreads();
  for (int w = 128; w > 0; w >>= 1){ if (t < w) red[t] = fmaxf(red[t], red[t+w]); __syncthreads(); }
  mx = red[0]; __syncthreads();
  float sum = 0.f;
  for (int j = t; j < n; j += 256){ float e = __expf((float)r[j] - mx); sum += e; r[j] = (f16)e; }
  red[t] = sum; __syncthreads();
  for (int w = 128; w > 0; w >>= 1){ if (t < w) red[t] += red[t+w]; __syncthreads(); }
  float inv = 1.f/red[0];
  for (int j = t; j < n; j += 256) r[j] = (f16)((float)r[j]*inv);
}

__global__ __launch_bounds__(256)
void k_layernorm(const float* __restrict__ X, const float* __restrict__ g,
                 const float* __restrict__ b, f16* __restrict__ out, int D){
  long long row = blockIdx.x;
  const float* x = X + row*(long long)D;
  __shared__ float red[256];
  int t = threadIdx.x;
  float s = 0.f; for (int j = t; j < D; j += 256) s += x[j];
  red[t] = s; __syncthreads();
  for (int w = 128; w > 0; w >>= 1){ if (t < w) red[t] += red[t+w]; __syncthreads(); }
  float mean = red[0]/D; __syncthreads();
  float v = 0.f; for (int j = t; j < D; j += 256){ float d = x[j]-mean; v += d*d; }
  red[t] = v; __syncthreads();
  for (int w = 128; w > 0; w >>= 1){ if (t < w) red[t] += red[t+w]; __syncthreads(); }
  float inv = rsqrtf(red[0]/D + 1e-5f);
  for (int j = t; j < D; j += 256)
    out[row*(long long)D + j] = (f16)((x[j]-mean)*inv*g[j] + b[j]);
}

__global__ void k_meanpool(const f16* __restrict__ t, float* __restrict__ out,
                           int B, int S, int D){
  int i = blockIdx.x*blockDim.x + threadIdx.x;
  if (i >= B*D) return;
  int b = i/D, d = i%D;
  float s = 0.f;
  for (int j = 0; j < S; ++j) s += (float)t[((long long)b*S + j)*D + d];
  out[i] = s/(float)S;
}

__global__ __launch_bounds__(256)
void k_head(const float* __restrict__ pooled, const float* __restrict__ g,
            const float* __restrict__ b, const float* __restrict__ W,
            const float* __restrict__ bias, float* __restrict__ out, int D, int NC){
  int bz = blockIdx.x;
  const float* x = pooled + (long long)bz*D;
  __shared__ float red[256];
  __shared__ float ln[512];
  int t = threadIdx.x;
  float s = 0.f; for (int j = t; j < D; j += 256) s += x[j];
  red[t] = s; __syncthreads();
  for (int w = 128; w > 0; w >>= 1){ if (t < w) red[t] += red[t+w]; __syncthreads(); }
  float mean = red[0]/D; __syncthreads();
  float v = 0.f; for (int j = t; j < D; j += 256){ float d = x[j]-mean; v += d*d; }
  red[t] = v; __syncthreads();
  for (int w = 128; w > 0; w >>= 1){ if (t < w) red[t] += red[t+w]; __syncthreads(); }
  float inv = rsqrtf(red[0]/D + 1e-5f);
  for (int j = t; j < D; j += 256) ln[j] = (x[j]-mean)*inv*g[j] + b[j];
  __syncthreads();
  if (t < NC) {
    float acc = bias[t];
    for (int j = 0; j < D; ++j) acc += ln[j]*W[j*NC + t];
    out[bz*NC + t] = acc;
  }
}

// ---------------------------------------------------------------------------
// Host-side orchestration
// ---------------------------------------------------------------------------

static void gemm(hipStream_t st, const f16* A, const f16* Bp, float* Cf, f16* Ch,
                 const f16* res, const float* sc, const float* sh,
                 int M, int N, int K32, int lda, int ldc,
                 int flags, float cs, int rowsPerB,
                 int Z = 1, int zdiv = 1,
                 long long sAo = 0, long long sAi = 0,
                 long long sCo = 0, long long sCi = 0)
{
  int tiles = (M/16)*(N/32);
  dim3 grid(cdiv(tiles, 8), 1, Z);
  k_gemm<<<grid, 256, 0, st>>>(A, Bp, Cf, Ch, res, sc, sh, M, N, K32, lda, ldc,
                               flags, cs, rowsPerB, zdiv, sAo, sAi, sCo, sCi);
}

struct Unit {
  const float *W, *b, *beta, *gamma, *mean, *var;
  int Cin, Cout, k, s, pad, Kpad;
  f16* Bp; float *scale, *shift, *fill;
};

static void prep_unit(hipStream_t st, const Unit& u){
  int KK = u.k*u.k, K32 = u.Kpad/32;
  long long tot = (long long)(u.Cout/16)*K32*512;
  k_pack_conv<<<cdiv(tot,256),256,0,st>>>(u.W, u.Bp, u.Cin, u.Cout, KK, K32);
  k_bnprep<<<cdiv(u.Cout,256),256,0,st>>>(u.b, u.gamma, u.beta, u.mean, u.var,
                                          u.scale, u.shift, u.fill, u.Cout);
}

static void run_unit(hipStream_t st, const Unit& u, const f16* in_feat,
                     f16* out_feat, f16* Abuf){
  const int B = 8, H = 32, W = 32;
  int P1 = (u.s == 1) ? 32 : 16;
  int M = B*P1*P1, Kreal = u.Cin*u.k*u.k;
  if (u.s > 1) {
    int nf = B*(1024 - P1*P1)*u.Cout;
    k_fill<<<cdiv(nf,256),256,0,st>>>(out_feat, u.fill, B, u.Cout, 1024, P1*P1);
  }
  long long tot = (long long)M*u.Kpad;
  k_im2col<<<cdiv(tot,256),256,0,st>>>(in_feat, Abuf, B, u.Cin, H, W,
                                       u.k, u.s, u.pad, P1, Kreal, u.Kpad);
  gemm(st, Abuf, u.Bp, nullptr, out_feat, nullptr, u.scale, u.shift,
       M, u.Cout, u.Kpad/32, u.Kpad, u.Cout, EP_BN | EP_RELU, 0.f,
       (P1*P1 == 1024) ? 0 : P1*P1);
}

static void run_mha(hipStream_t st, const f16* t,
                    const f16* Wqp, const f16* Wkp, const f16* Wvp, const f16* Wop,
                    const float* g, const float* b,
                    f16* q, f16* kb, f16* v, f16* kpk, f16* vpk,
                    f16* scores, f16* ctx, float* lnin, f16* tout)
{
  const int B = 8, S = 1024, D = 512, H = 8, DK = 64, M = B*S;
  gemm(st, t, Wqp, nullptr, q,  nullptr, nullptr, nullptr, M, D, 16, D, D, 0, 0.f, 0);
  gemm(st, t, Wkp, nullptr, kb, nullptr, nullptr, nullptr, M, D, 16, D, D, 0, 0.f, 0);
  gemm(st, t, Wvp, nullptr, v,  nullptr, nullptr, nullptr, M, D, 16, D, D, 0, 0.f, 0);
  // pack K^T (64 x 1024 per z) and V (1024 x 64 per z), z = b*H + h
  long long per_z = 65536; // (N/16)*K32*512 for both
  k_pack_f16<<<cdiv(per_z*B*H,256),256,0,st>>>(kb, kpk, DK, S, 2, D, 1,
                                               B*H, H, (long long)S*D, DK);
  k_pack_f16<<<cdiv(per_z*B*H,256),256,0,st>>>(v, vpk, S, DK, 32, D, 0,
                                               B*H, H, (long long)S*D, DK);
  // scores = (Q_bh/sqrt(dk)) K_bh^T
  gemm(st, q, kpk, nullptr, scores, nullptr, nullptr, nullptr,
       S, S, 2, D, S, EP_CSCALE, 0.125f, 0,
       B*H, H, (long long)S*D, DK, (long long)H*S*S, (long long)S*S);
  k_softmax<<<dim3(B*H*S), 256, 0, st>>>(scores, S);
  // ctx = P V, written interleaved into [M, D] at column h*DK
  gemm(st, scores, vpk, nullptr, ctx, nullptr, nullptr, nullptr,
       S, DK, 32, S, D, 0, 0.f, 0,
       B*H, H, (long long)H*S*S, (long long)S*S, (long long)S*D, DK);
  // out = ctx @ Wo + t (f32), then LayerNorm -> f16
  gemm(st, ctx, Wop, lnin, nullptr, t, nullptr, nullptr,
       M, D, 16, D, D, EP_RESID, 0.f, 0);
  k_layernorm<<<dim3(M), 256, 0, st>>>(lnin, g, b, tout, D);
}

extern "C" void kernel_launch(void* const* d_in, const int* in_sizes, int n_in,
                              void* d_out, int out_size, void* d_ws, size_t ws_size,
                              hipStream_t stream)
{
  (void)out_size; (void)ws_size;
  // JAX pytree leaf order (dicts flattened with sorted keys). Detect whether the
  // harness put `x` (8*3*32*32 = 24576 elems) first or last.
  int pbase, xi;
  if (in_sizes[0] == 24576) { xi = 0; pbase = 1; }
  else { xi = n_in - 1; pbase = 0; }
  const float* X = (const float*)d_in[xi];
  auto P = [&](int i){ return (const float*)d_in[pbase + i]; };
  // sorted param leaves:
  // 0 ffn.W1  1 ffn.W2  2 ffn.b  3 ffn.g
  // 4 m1.Wk 5 m1.Wo 6 m1.Wq 7 m1.Wv 8 m1.b 9 m1.g ; 10..15 mha2 likewise
  // 16 proj.W 17 proj.b 18 proj.bias 19 proj.g
  // 20..25 sc1{W,b,beta,gamma,mean,var} 26..31 sc2
  // 32..37 u1a 38..43 u1b 44..49 u1c 50..55 u2a 56..61 u2b 62..67 u3

  char* wp = (char*)d_ws;
  auto alloc = [&](size_t bytes)->void*{
    void* r = (void*)wp; wp += (bytes + 255) & ~(size_t)255; return r;
  };

  // activations (feature matrices [8192, C], f16)
  f16* featX = (f16*)alloc((size_t)8192*3*2);
  f16* fu1a  = (f16*)alloc((size_t)8192*64*2);
  f16* fu1b  = (f16*)alloc((size_t)8192*128*2);
  f16* fu1c  = (f16*)alloc((size_t)8192*256*2);
  f16* fsc1  = (f16*)alloc((size_t)8192*256*2);
  f16* fx1   = (f16*)alloc((size_t)8192*256*2);
  f16* fu2a  = (f16*)alloc((size_t)8192*512*2);
  f16* fu2b  = (f16*)alloc((size_t)8192*512*2);
  f16* fsc2  = (f16*)alloc((size_t)8192*512*2);
  f16* fx2   = (f16*)alloc((size_t)8192*512*2);
  f16* fx3   = (f16*)alloc((size_t)8192*512*2);
  f16* qb    = (f16*)alloc((size_t)8192*512*2);
  f16* kbuf  = (f16*)alloc((size_t)8192*512*2);
  f16* vb    = (f16*)alloc((size_t)8192*512*2);
  f16* kpk   = (f16*)alloc((size_t)8192*512*2);   // packed K^T blocks
  f16* vpk   = (f16*)alloc((size_t)8192*512*2);   // packed V blocks
  f16* ctx   = (f16*)alloc((size_t)8192*512*2);
  f16* t1    = (f16*)alloc((size_t)8192*512*2);
  f16* t2    = (f16*)alloc((size_t)8192*512*2);
  f16* t3    = (f16*)alloc((size_t)8192*512*2);
  // big region (time-shared): im2col A (38MB) / attention scores (128MB) / FFN hidden (32MB)
  char* big   = (char*)alloc((size_t)64*1024*1024*2);
  f16* Abuf   = (f16*)big;
  f16* scores = (f16*)big;
  f16* ffnh   = (f16*)big;
  float* lnin   = (float*)alloc((size_t)8192*512*4);
  float* pooled = (float*)alloc((size_t)8*512*4);

  // dense weights -> packed f16
  auto packDense = [&](const float* src, int K, int N)->f16*{
    int K32 = (K + 31)/32;
    long long tot = (long long)(N/16)*K32*512;
    f16* d = (f16*)alloc((size_t)tot*2);
    k_pack_dense<<<cdiv(tot,256),256,0,stream>>>(src, d, K, N, K32);
    return d;
  };
  f16* m1wk = packDense(P(4),  512, 512); f16* m1wo = packDense(P(5),  512, 512);
  f16* m1wq = packDense(P(6),  512, 512); f16* m1wv = packDense(P(7),  512, 512);
  f16* m2wk = packDense(P(10), 512, 512); f16* m2wo = packDense(P(11), 512, 512);
  f16* m2wq = packDense(P(12), 512, 512); f16* m2wv = packDense(P(13), 512, 512);
  f16* w1p  = packDense(P(0),  512, 2048);
  f16* w2p  = packDense(P(1),  2048, 512);

  auto mkunit = [&](int pi, int Cin, int Cout, int k, int s, int pad)->Unit{
    Unit u;
    u.W = P(pi); u.b = P(pi+1); u.beta = P(pi+2); u.gamma = P(pi+3);
    u.mean = P(pi+4); u.var = P(pi+5);
    u.Cin = Cin; u.Cout = Cout; u.k = k; u.s = s; u.pad = pad;
    u.Kpad = ((Cin*k*k + 31)/32)*32;
    u.Bp = (f16*)alloc((size_t)(Cout/16)*(u.Kpad/32)*512*2);
    u.scale = (float*)alloc((size_t)Cout*4);
    u.shift = (float*)alloc((size_t)Cout*4);
    u.fill  = (float*)alloc((size_t)Cout*4);
    return u;
  };
  Unit usc1 = mkunit(20, 3,   256, 1, 2, 0);
  Unit usc2 = mkunit(26, 256, 512, 1, 2, 0);
  Unit uu1a = mkunit(32, 3,   64,  3, 1, 1);
  Unit uu1b = mkunit(38, 64,  128, 3, 2, 1);
  Unit uu1c = mkunit(44, 128, 256, 3, 1, 1);
  Unit uu2a = mkunit(50, 256, 512, 3, 1, 1);
  Unit uu2b = mkunit(56, 512, 512, 3, 2, 1);
  Unit uu3  = mkunit(62, 512, 512, 3, 2, 1);

  prep_unit(stream, usc1); prep_unit(stream, usc2);
  prep_unit(stream, uu1a); prep_unit(stream, uu1b); prep_unit(stream, uu1c);
  prep_unit(stream, uu2a); prep_unit(stream, uu2b); prep_unit(stream, uu3);

  // ---- forward pass ----
  k_nchw_to_feat<<<cdiv(8*3*1024,256),256,0,stream>>>(X, featX, 8, 3, 1024);

  run_unit(stream, uu1a, featX, fu1a, Abuf);
  run_unit(stream, uu1b, fu1a,  fu1b, Abuf);
  run_unit(stream, uu1c, fu1b,  fu1c, Abuf);
  run_unit(stream, usc1, featX, fsc1, Abuf);
  k_addrelu<<<cdiv(8192*256,256),256,0,stream>>>(fsc1, fu1c, fx1, 8192*256);

  run_unit(stream, uu2a, fx1,  fu2a, Abuf);
  run_unit(stream, uu2b, fu2a, fu2b, Abuf);
  run_unit(stream, usc2, fx1,  fsc2, Abuf);
  k_addrelu<<<cdiv(8192*512,256),256,0,stream>>>(fsc2, fu2b, fx2, 8192*512);

  run_unit(stream, uu3, fx2, fx3, Abuf);      // fx3 == t : [B=8, S=1024, D=512]

  run_mha(stream, fx3, m1wq, m1wk, m1wv, m1wo, P(9), P(8),
          qb, kbuf, vb, kpk, vpk, scores, ctx, lnin, t1);
  run_mha(stream, t1,  m2wq, m2wk, m2wv, m2wo, P(15), P(14),
          qb, kbuf, vb, kpk, vpk, scores, ctx, lnin, t2);

  // FFN: relu(t2 @ W1) @ W2 + t2 -> LN
  gemm(stream, t2, w1p, nullptr, ffnh, nullptr, nullptr, nullptr,
       8192, 2048, 16, 512, 2048, EP_RELU, 0.f, 0);
  gemm(stream, ffnh, w2p, lnin, nullptr, t2, nullptr, nullptr,
       8192, 512, 64, 2048, 512, EP_RESID, 0.f, 0);
  k_layernorm<<<dim3(8192),256,0,stream>>>(lnin, P(3), P(2), t3, 512);

  // mean over sequence -> LN -> proj
  k_meanpool<<<cdiv(8*512,256),256,0,stream>>>(t3, pooled, 8, 1024, 512);
  k_head<<<dim3(8),256,0,stream>>>(pooled, P(19), P(17), P(16), P(18),
                                   (float*)d_out, 512, 10);
}
